// CustomLSTM_78271484002847
// MI455X (gfx1250) — matrix operand order repaired and checked by reading, not compiled, not run on hardware
//
#include <hip/hip_runtime.h>
#include <hip/hip_bf16.h>

// ---------------------------------------------------------------------------
// CDNA5 (gfx1250, wave32) LSTM:
//   * all GEMMs via v_wmma_f32_16x16x32_bf16
//   * weight fragments staged WG-wide into LDS by the Tensor Data Mover
//     (tensor_load_to_lds + s_wait_tensorcnt), double-buffered
//   * each wave processes 2 batch tiles so every LDS B-fragment read feeds
//     2 WMMAs; all loads of a round are hoisted ahead of the WMMA burst.
// ---------------------------------------------------------------------------

typedef __attribute__((ext_vector_type(16))) __bf16       v16bf;
typedef __attribute__((ext_vector_type(8)))  __bf16       v8bf;
typedef __attribute__((ext_vector_type(8)))  float        v8f;
typedef __attribute__((ext_vector_type(4)))  unsigned int u32x4;
typedef __attribute__((ext_vector_type(8)))  unsigned int u32x8;

#define LSTM_B 256
#define LSTM_T 128
#define LSTM_F 128
#define LSTM_H 1024
#define LSTM_O 128
#define LSTM_TH (LSTM_T * LSTM_H)   // 131072

__device__ __forceinline__ float sig_f(float x)  { return 1.0f / (1.0f + __expf(-x)); }
__device__ __forceinline__ float tanh_f(float x) {
    float e = __expf(2.0f * x);
    return (e - 1.0f) / (e + 1.0f);
}

// A-fragment (16x32, MxK) per the CDNA5 ISA layout:
//   lanes 0-15 : row = rowBase + lane,    K = kt*32 + {0..7, 16..23}
//   lanes 16-31: row = rowBase + lane-16, K = kt*32 + {8..15, 24..31}
__device__ __forceinline__ v16bf load_a_frag(const __bf16* __restrict__ base,
                                             int rowStride, int rowBase,
                                             int kt, int lane) {
    int row = rowBase + (lane & 15);
    int kb  = (kt << 5) + ((lane & 16) ? 8 : 0);
    const __bf16* p = base + (size_t)row * rowStride + kb;
    v8bf lo = *(const v8bf*)p;
    v8bf hi = *(const v8bf*)(p + 16);
    v16bf a;
#pragma unroll
    for (int e = 0; e < 8; ++e) { a[e] = lo[e]; a[8 + e] = hi[e]; }
    return a;
}

// TDM 2-D copy: `rows` rows of 2048 bytes each (gap `rowStrideBytes` between
// rows in global memory) -> contiguous 8KB block at LDS offset `ldsOff`.
// D# group0/group1 built per CDNA5 ISA ch.8; groups 2/3 omitted (<=2D tensor).
__device__ __forceinline__ void tdm_copy_rows(unsigned int ldsOff, const void* gsrc,
                                              unsigned int rows,
                                              unsigned long long rowStrideBytes) {
    unsigned long long ga = (unsigned long long)(uintptr_t)gsrc;
    u32x4 g0;
    g0[0] = 1u;                                             // count=1, no gather
    g0[1] = ldsOff;                                         // lds_addr
    g0[2] = (unsigned int)ga;                               // global_addr[31:0]
    g0[3] = ((unsigned int)(ga >> 32) & 0x01FFFFFFu)        // global_addr[56:32]
            | 0x80000000u;                                  // type=2 (image)
    const unsigned int d0 = 2048u >> 3;                     // row length, 8B units
    unsigned long long s8 = rowStrideBytes >> 3;            // row stride, 8B units
    u32x8 g1;
    g1[0] = 3u << 16;                                       // data_size=8B, mask=0
    g1[1] = (d0 & 0xFFFFu) << 16;                           // tensor_dim0 lo16
    g1[2] = (d0 >> 16) | ((rows & 0xFFFFu) << 16);          // dim0 hi16 | dim1 lo16
    g1[3] = ((rows >> 16) & 0xFFFFu) | (d0 << 16);          // dim1 hi16 | tile_dim0
    g1[4] = rows & 0xFFFFu;                                 // tile_dim1 | tile_dim2=0
    g1[5] = (unsigned int)s8;                               // dim0_stride lo32
    g1[6] = (unsigned int)(s8 >> 32) & 0xFFFFu;             // dim0_stride hi16
    g1[7] = 0u;
    asm volatile("tensor_load_to_lds %0, %1" :: "s"(g0), "s"(g1) : "memory");
}

// Shared GEMM phase for one hidden-column tile jt:
//   acc[bt2][g] += act(tile w, w+8) @ W_g^T, g in {i,f,g,o}.
// Weights TDM-staged into LDS (double-buffered, 2 k-tiles x 4 gates = 8KB per
// round) and shared by the 8 waves; each B-fragment register feeds 2 WMMAs.
__device__ __forceinline__ void gemm_phase(const __bf16* __restrict__ act, int actStride,
                                           const __bf16* __restrict__ wf, int nkt,
                                           int jt, int wave, int lane,
                                           __bf16* smem, v8f (*acc)[4]) {
    const int nrounds = nkt >> 1;
    unsigned long long gateStrideB = (unsigned long long)(64 * nkt) * 1024ull; // bytes
    const __bf16* gbase = wf + (((size_t)jt * nkt) << 9);   // gate0, kt=0 fragment
    unsigned int ldsBase = (unsigned int)(uintptr_t)smem;

    if (wave == 0) {
        tdm_copy_rows(ldsBase, gbase, 4, gateStrideB);
        __builtin_amdgcn_s_wait_tensorcnt(0);
    }
    __syncthreads();

    int buf = 0;
    for (int r = 0; r < nrounds; ++r) {
        bool more = (r + 1) < nrounds;
        if (wave == 0 && more)
            tdm_copy_rows(ldsBase + ((unsigned)(buf ^ 1) << 13),
                          gbase + ((size_t)(r + 1) << 10), 4, gateStrideB);
        const __bf16* sb = smem + (buf << 12);

        // ---- hoist ALL loads of this round, then issue the WMMA burst ----
        v16bf a[2][2];                              // [btile-half][ktSub]
#pragma unroll
        for (int b2 = 0; b2 < 2; ++b2)
#pragma unroll
            for (int s = 0; s < 2; ++s)
                a[b2][s] = load_a_frag(act, actStride, (wave + (b2 << 3)) << 4,
                                       2 * r + s, lane);
        v16bf bf[2][4];                             // [ktSub][gate]
#pragma unroll
        for (int s = 0; s < 2; ++s)
#pragma unroll
            for (int g = 0; g < 4; ++g)
                bf[s][g] = *(const v16bf*)(sb + (g << 10) + (s << 9) + (lane << 4));

#pragma unroll
        for (int s = 0; s < 2; ++s)
#pragma unroll
            for (int g = 0; g < 4; ++g) {
#pragma unroll
                for (int b2 = 0; b2 < 2; ++b2)
                    acc[b2][g] = __builtin_amdgcn_wmma_f32_16x16x32_bf16(
                        false, a[b2][s], false, bf[s][g], (short)0,
                        acc[b2][g], false, false);
            }

        if (wave == 0 && more) __builtin_amdgcn_s_wait_tensorcnt(0);
        __syncthreads();
        buf ^= 1;
    }
}

// Pack W [N x K] (f32 row-major; GEMM uses W^T as B) into fragment-major bf16:
// fragment (nt,kt) = 32x16 tile in per-lane WMMA B layout, 1KB, kt-contiguous.
__global__ __launch_bounds__(256) void pack_b_frags(const float* __restrict__ W,
                                                    __bf16* __restrict__ out,
                                                    int N, int K) {
    int lane = threadIdx.x & 31;
    int frag = blockIdx.x * (blockDim.x >> 5) + (threadIdx.x >> 5);
    int nkt = K >> 5;
    if (frag >= (N >> 4) * nkt) return;
    int nt = frag / nkt, kt = frag - nt * nkt;
    int n  = (nt << 4) + (lane & 15);
    int kb = (kt << 5) + ((lane >> 4) << 3);
    const float* src = W + (size_t)n * K + kb;
    __bf16* dst = out + (((size_t)frag) << 9) + (lane << 4);
#pragma unroll
    for (int e = 0; e < 16; ++e) dst[e] = (__bf16)src[e < 8 ? e : e + 8];
}

__global__ __launch_bounds__(256) void cvt_f32_bf16(const float* __restrict__ in,
                                                    __bf16* __restrict__ out, int n) {
    int i = blockIdx.x * blockDim.x + threadIdx.x;
    if (i < n) out[i] = (__bf16)in[i];
}

__global__ __launch_bounds__(256) void combine_bias(const float* __restrict__ a,
                                                    const float* __restrict__ b,
                                                    float* __restrict__ o, int n) {
    int i = blockIdx.x * blockDim.x + threadIdx.x;
    if (i < n) o[i] = a[i] + b[i];
}

// One LSTM layer step.  Grid: 64 blocks (one per hidden-column tile jt) of
// 256 threads = 8 waves; wave w owns batch tiles w and w+8, fusing the 4
// gate tiles of each so the nonlinearity happens fully in registers.
__global__ __launch_bounds__(256) void lstm_step(
    const __bf16* __restrict__ x, int Kin,          // [B x Kin] row-major bf16
    const __bf16* __restrict__ hprev,               // [B x H]   row-major bf16
    const __bf16* __restrict__ wih_frag,            // [4H x Kin] fragment-packed
    const __bf16* __restrict__ whh_frag,            // [4H x H]  fragment-packed
    const float*  __restrict__ bias,                // [4H] = bih + bhh
    float*        __restrict__ cstate,              // [B x H] f32, in-place
    __bf16*       __restrict__ hout,                // [B x H] bf16 (next-step buf)
    __bf16*       __restrict__ outs_slot)           // staging slot or nullptr
{
    __shared__ __attribute__((aligned(16))) __bf16 smem[2 * 4 * 2 * 512]; // 16KB
    const int H = LSTM_H;
    int lane = threadIdx.x & 31;
    int wave = threadIdx.x >> 5;     // 0..7; owns batch tiles wave, wave+8
    int jt   = blockIdx.x;           // hidden-column tile 0..63

    v8f acc[2][4];
#pragma unroll
    for (int b2 = 0; b2 < 2; ++b2)
#pragma unroll
        for (int g = 0; g < 4; ++g) acc[b2][g] = v8f{};

    gemm_phase(x,     Kin, wih_frag, Kin >> 5, jt, wave, lane, smem, acc); // x @ Wih^T
    gemm_phase(hprev, H,   whh_frag, H  >> 5, jt, wave, lane, smem, acc);  // h @ Whh^T

    // C/D layout: N = lane&15 (hidden), M = (lane>>4)*8 + r (batch row).
    int h = (jt << 4) + (lane & 15);
    float bi = bias[h];
    float bf = bias[H + h];
    float bg = bias[2 * H + h];
    float bo = bias[3 * H + h];
#pragma unroll
    for (int b2 = 0; b2 < 2; ++b2) {
        int rbase = ((wave + (b2 << 3)) << 4) + ((lane >> 4) << 3);
#pragma unroll
        for (int r = 0; r < 8; ++r) {
            size_t idx = (size_t)(rbase + r) * H + h;
            float iv = sig_f(acc[b2][0][r] + bi);
            float fv = sig_f(acc[b2][1][r] + bf);
            float gv = tanh_f(acc[b2][2][r] + bg);
            float ov = sig_f(acc[b2][3][r] + bo);
            float cn = fv * cstate[idx] + iv * gv;
            cstate[idx] = cn;
            float hn = ov * tanh_f(cn);
            __bf16 hb = (__bf16)hn;
            hout[idx] = hb;
            if (outs_slot) outs_slot[idx] = hb;
        }
    }
}

// Final GEMM: out[256 x 128] = flat[256 x 131072] @ Wout^T + bout.
// flat is the pure-reshape view of outs [T,B,H]; h1 was staged in that exact
// linear order.  Grid: 128 waves = 16 blocks x 8.
__global__ __launch_bounds__(256) void out_gemm(
    const __bf16* __restrict__ outs,      // [B x TH] row-major bf16
    const __bf16* __restrict__ wout_frag, // [O x TH] fragment-packed
    const float*  __restrict__ bout,
    float*        __restrict__ out)
{
    int lane = threadIdx.x & 31;
    int gw   = blockIdx.x * (blockDim.x >> 5) + (threadIdx.x >> 5);
    int ot   = gw & 7;            // O/16 = 8
    int bt   = gw >> 3;           // B/16 = 16
    int rowBase = bt << 4;

    v8f acc = v8f{};
    const int nkt = LSTM_TH >> 5; // 4096
    for (int kt = 0; kt < nkt; ++kt) {
        v16bf a = load_a_frag(outs, LSTM_TH, rowBase, kt, lane);
        const v16bf* bp = (const v16bf*)(wout_frag + ((((size_t)ot * nkt + kt) << 9) + (lane << 4)));
        acc = __builtin_amdgcn_wmma_f32_16x16x32_bf16(
            false, a, false, *bp, (short)0, acc, false, false);
    }
    int o     = (ot << 4) + (lane & 15);
    int rbase = rowBase + ((lane >> 4) << 3);
    float bv = bout[o];
#pragma unroll
    for (int r = 0; r < 8; ++r)
        out[(size_t)(rbase + r) * LSTM_O + o] = acc[r] + bv;
}

// ---------------------------------------------------------------------------

extern "C" void kernel_launch(void* const* d_in, const int* in_sizes, int n_in,
                              void* d_out, int out_size, void* d_ws, size_t ws_size,
                              hipStream_t stream) {
    (void)in_sizes; (void)n_in; (void)out_size; (void)ws_size;
    const float* batch = (const float*)d_in[0];
    const float* Wih0  = (const float*)d_in[1];
    const float* Whh0  = (const float*)d_in[2];
    const float* bih0  = (const float*)d_in[3];
    const float* bhh0  = (const float*)d_in[4];
    const float* Wih1  = (const float*)d_in[5];
    const float* Whh1  = (const float*)d_in[6];
    const float* bih1  = (const float*)d_in[7];
    const float* bhh1  = (const float*)d_in[8];
    const float* h00   = (const float*)d_in[9];
    const float* c00   = (const float*)d_in[10];
    const float* h01   = (const float*)d_in[11];
    const float* c01   = (const float*)d_in[12];
    const float* Wout  = (const float*)d_in[13];
    const float* bout  = (const float*)d_in[14];

    char* ws = (char*)d_ws;
    size_t off = 0;
    auto take = [&](size_t bytes) -> char* {
        char* p = ws + off;
        off += (bytes + 255) & ~(size_t)255;
        return p;
    };

    const size_t BH = (size_t)LSTM_B * LSTM_H;
    __bf16* xall  = (__bf16*)take((size_t)LSTM_T * LSTM_B * LSTM_F * 2); // 8.4 MB
    __bf16* wih0f = (__bf16*)take((size_t)4 * LSTM_H * LSTM_F * 2);      // 1 MB
    __bf16* whh0f = (__bf16*)take((size_t)4 * LSTM_H * LSTM_H * 2);      // 8.4 MB
    __bf16* wih1f = (__bf16*)take((size_t)4 * LSTM_H * LSTM_H * 2);      // 8.4 MB
    __bf16* whh1f = (__bf16*)take((size_t)4 * LSTM_H * LSTM_H * 2);      // 8.4 MB
    __bf16* woutf = (__bf16*)take((size_t)LSTM_O * LSTM_TH * 2);         // 33.5 MB
    float*  b0    = (float*)take((size_t)4 * LSTM_H * 4);
    float*  b1    = (float*)take((size_t)4 * LSTM_H * 4);
    __bf16* h0buf[2] = {(__bf16*)take(BH * 2), (__bf16*)take(BH * 2)};
    __bf16* h1buf[2] = {(__bf16*)take(BH * 2), (__bf16*)take(BH * 2)};
    float*  c0    = (float*)take(BH * 4);
    float*  c1    = (float*)take(BH * 4);
    __bf16* outs  = (__bf16*)take((size_t)LSTM_T * BH * 2);              // 67 MB

    // ---- pre-pass: pack weights, biases, batch, state
    auto packBlocks = [](int N, int K) { return ((N >> 4) * (K >> 5) + 7) / 8; };
    pack_b_frags<<<packBlocks(4 * LSTM_H, LSTM_F), 256, 0, stream>>>(Wih0, wih0f, 4 * LSTM_H, LSTM_F);
    pack_b_frags<<<packBlocks(4 * LSTM_H, LSTM_H), 256, 0, stream>>>(Whh0, whh0f, 4 * LSTM_H, LSTM_H);
    pack_b_frags<<<packBlocks(4 * LSTM_H, LSTM_H), 256, 0, stream>>>(Wih1, wih1f, 4 * LSTM_H, LSTM_H);
    pack_b_frags<<<packBlocks(4 * LSTM_H, LSTM_H), 256, 0, stream>>>(Whh1, whh1f, 4 * LSTM_H, LSTM_H);
    pack_b_frags<<<packBlocks(LSTM_O, LSTM_TH), 256, 0, stream>>>(Wout, woutf, LSTM_O, LSTM_TH);

    int nbatch = LSTM_T * LSTM_B * LSTM_F;
    cvt_f32_bf16<<<(nbatch + 255) / 256, 256, 0, stream>>>(batch, xall, nbatch);
    combine_bias<<<(4 * LSTM_H + 255) / 256, 256, 0, stream>>>(bih0, bhh0, b0, 4 * LSTM_H);
    combine_bias<<<(4 * LSTM_H + 255) / 256, 256, 0, stream>>>(bih1, bhh1, b1, 4 * LSTM_H);
    cvt_f32_bf16<<<((int)BH + 255) / 256, 256, 0, stream>>>(h00, h0buf[0], (int)BH);
    cvt_f32_bf16<<<((int)BH + 255) / 256, 256, 0, stream>>>(h01, h1buf[0], (int)BH);
    hipMemcpyAsync(c0, c00, BH * sizeof(float), hipMemcpyDeviceToDevice, stream);
    hipMemcpyAsync(c1, c01, BH * sizeof(float), hipMemcpyDeviceToDevice, stream);

    // ---- recurrence: 2 kernels / step, h double-buffered across steps
    for (int t = 0; t < LSTM_T; ++t) {
        int cur = t & 1, nxt = cur ^ 1;
        const __bf16* xt = xall + (size_t)t * LSTM_B * LSTM_F;  // reshape(T,B,F)[t]
        lstm_step<<<64, 256, 0, stream>>>(xt, LSTM_F, h0buf[cur],
                                          wih0f, whh0f, b0, c0,
                                          h0buf[nxt], (__bf16*)nullptr);
        lstm_step<<<64, 256, 0, stream>>>(h0buf[nxt], LSTM_H, h1buf[cur],
                                          wih1f, whh1f, b1, c1,
                                          h1buf[nxt], outs + (size_t)t * BH);
    }

    // ---- output projection
    out_gemm<<<16, 256, 0, stream>>>(outs, woutf, bout, (float*)d_out);
}
